// TimesFmAttention_72121090834491
// MI455X (gfx1250) — matrix operand order repaired and checked by reading, not compiled, not run on hardware
//
#include <hip/hip_runtime.h>
#include <math.h>

// TimesFM attention for MI455X (gfx1250, wave32, WMMA bf16 16x16x32, f32 accum).
// B=4 S=2048 D=1280 H=16 HD=80. Head dim padded to 96 (3 x K=32 WMMA steps).
// attention_mask input (d_in[1]) is the causal -1e9 mask; implemented analytically.
// GEMMs use a register-staged double-buffer pipeline: next K-tile's global
// loads are issued right after the fill barrier so they overlap ds_load+wmma.

#define S_LEN   2048
#define DMODEL  1280
#define NHEAD   16
#define HDIM    80
#define HDPAD   96
#define BATCH   4
#define MTOT    (BATCH * S_LEN)     // 8192
#define NQKV    (3 * NHEAD * HDIM)  // 3840

typedef __bf16 bf16;
typedef __attribute__((ext_vector_type(8)))  __bf16 v8bf;
typedef __attribute__((ext_vector_type(16))) __bf16 v16bf;
typedef __attribute__((ext_vector_type(8)))  float  v8f;

static __device__ __forceinline__ v16bf join16(v8bf a, v8bf b) {
  v16bf r;
#pragma unroll
  for (int i = 0; i < 8; ++i) { r[i] = a[i]; r[i + 8] = b[i]; }
  return r;
}

static __device__ __forceinline__ v8f wmma_bf16(v16bf a, v16bf b, v8f c) {
  // (neg_a, A, neg_b, B, c_mod, C, reuse_a, reuse_b)
  return __builtin_amdgcn_wmma_f32_16x16x32_bf16(false, a, false, b, (short)0, c,
                                                 false, false);
}

// A fragment (16x32 bf16). ISA 7.12.2: lane&15 = row M; lane>>4 selects
// k-halves {0..7,16..23} (hi=0) vs {8..15,24..31} (hi=1). Two b128 loads.
static __device__ __forceinline__ v16bf frag_a(const bf16* p, int hi) {
  v8bf x = *(const v8bf*)(p + hi * 8);
  v8bf y = *(const v8bf*)(p + 16 + hi * 8);
  return join16(x, y);
}

// B fragment (32x16 bf16). lane&15 = col N; lane>>4 selects k-halves
// {0..7,16..23} (hi=0) vs {8..15,24..31} (hi=1) => two b128 loads from an [n][k] row.
static __device__ __forceinline__ v16bf frag_b(const bf16* p, int hi) {
  v8bf x = *(const v8bf*)(p + hi * 8);
  v8bf y = *(const v8bf*)(p + 16 + hi * 8);
  return join16(x, y);
}

__global__ void k_cvt_bf16(const float* __restrict__ src, bf16* __restrict__ dst,
                           int n) {
  for (int i = blockIdx.x * blockDim.x + threadIdx.x; i < n;
       i += gridDim.x * blockDim.x)
    dst[i] = (bf16)src[i];
}

__global__ void k_zero_bf16(bf16* __restrict__ dst, int n8) {
  int i = blockIdx.x * blockDim.x + threadIdx.x;
  if (i < n8) { v8bf z = {}; *(v8bf*)(dst + (size_t)i * 8) = z; }
}

// ---------------- QKV projection: C = A @ W^T + b, scattered to Q/K/V ------
__global__ __launch_bounds__(256) void k_qkv_gemm(
    const bf16* __restrict__ A,        // [MTOT][DMODEL]
    const bf16* __restrict__ W,        // [NQKV][DMODEL]
    const float* __restrict__ bias,    // [NQKV]
    const float* __restrict__ scaling, // [HDIM]
    bf16* __restrict__ Qs,             // [B,H,S,HDPAD], pre-scaled
    bf16* __restrict__ Ks,             // [B,H,S,HDPAD]
    bf16* __restrict__ Vt)             // [B,H,HDIM,S]  (transposed)
{
  __shared__ bf16 lA[64 * 32];   // [m][k]
  __shared__ bf16 lB[128 * 32];  // [n][k]
  const int tid = threadIdx.x;
  const int lane = tid & 31, wid = tid >> 5;
  const int lo = lane & 15, hi = lane >> 4;
  const int wr = wid >> 2, wc = wid & 3;  // 2x4 wave grid
  const int m0 = blockIdx.y * 64;
  const int n0 = blockIdx.x * 128;

  // cooperative-fill indices (one A b128 + two B b128 per thread)
  const int mL = tid >> 2, segA = tid & 3;
  const int nL0 = tid >> 2,        segB0 = tid & 3;
  const int nL1 = (tid + 256) >> 2, segB1 = (tid + 256) & 3;
  const bf16* gA  = &A[(size_t)(m0 + mL) * DMODEL + segA * 8];
  const bf16* gB0 = &W[(size_t)(n0 + nL0) * DMODEL + segB0 * 8];
  const bf16* gB1 = &W[(size_t)(n0 + nL1) * DMODEL + segB1 * 8];

  v8f c[2][2] = {};
  v8bf ra  = *(const v8bf*)gA;          // stage k-tile 0
  v8bf rb0 = *(const v8bf*)gB0;
  v8bf rb1 = *(const v8bf*)gB1;

  for (int k0 = 0; k0 < DMODEL; k0 += 32) {
    *(v8bf*)&lA[mL * 32 + segA * 8]   = ra;
    *(v8bf*)&lB[nL0 * 32 + segB0 * 8] = rb0;
    *(v8bf*)&lB[nL1 * 32 + segB1 * 8] = rb1;
    __syncthreads();
    if (k0 + 32 < DMODEL) {            // issue next tile's loads now:
      ra  = *(const v8bf*)(gA + k0 + 32);   // overlap with ds_load + wmma
      rb0 = *(const v8bf*)(gB0 + k0 + 32);
      rb1 = *(const v8bf*)(gB1 + k0 + 32);
    }
    v16bf a0 = frag_a(&lA[(wr * 32 + 0 * 16 + lo) * 32], hi);
    v16bf a1 = frag_a(&lA[(wr * 32 + 1 * 16 + lo) * 32], hi);
    v16bf b0 = frag_b(&lB[(wc * 32 + 0 * 16 + lo) * 32], hi);
    v16bf b1 = frag_b(&lB[(wc * 32 + 1 * 16 + lo) * 32], hi);
    c[0][0] = wmma_bf16(a0, b0, c[0][0]);
    c[0][1] = wmma_bf16(a0, b1, c[0][1]);
    c[1][0] = wmma_bf16(a1, b0, c[1][0]);
    c[1][1] = wmma_bf16(a1, b1, c[1][1]);
    __syncthreads();
  }

  const float kScaleC = 0.16129823f;  // log2(e) / sqrt(80)
#pragma unroll
  for (int i = 0; i < 2; ++i)
#pragma unroll
    for (int j = 0; j < 2; ++j)
#pragma unroll
      for (int v = 0; v < 8; ++v) {
        int m = m0 + wr * 32 + i * 16 + v + hi * 8;  // C layout: row = v + hi*8
        int n = n0 + wc * 32 + j * 16 + lo;          // col = lane&15
        float val = c[i][j][v] + bias[n];
        int b = m >> 11, s = m & (S_LEN - 1);
        if (n < NHEAD * HDIM) {
          int h = n / HDIM, d = n - h * HDIM;
          float sc = scaling[d];
          float sp = (sc > 20.f) ? sc : log1pf(__expf(sc));  // softplus
          val *= sp * kScaleC;
          Qs[(((size_t)(b * NHEAD + h)) * S_LEN + s) * HDPAD + d] = (bf16)val;
        } else if (n < 2 * NHEAD * HDIM) {
          int nk = n - NHEAD * HDIM;
          int h = nk / HDIM, d = nk - h * HDIM;
          Ks[(((size_t)(b * NHEAD + h)) * S_LEN + s) * HDPAD + d] = (bf16)val;
        } else {
          int nv = n - 2 * NHEAD * HDIM;
          int h = nv / HDIM, d = nv - h * HDIM;
          Vt[(((size_t)(b * NHEAD + h)) * HDIM + d) * S_LEN + s] = (bf16)val;
        }
      }
}

// ---------------- Flash attention: one wave per 16 query rows --------------
__global__ __launch_bounds__(128) void k_flash_attn(
    const bf16* __restrict__ Qs, const bf16* __restrict__ Ks,
    const bf16* __restrict__ Vt, bf16* __restrict__ Oa /* [B,S,H*HD] */)
{
  __shared__ bf16 lP[4][16 * 32];  // wave-private P staging
  const int lane = threadIdx.x & 31, wid = threadIdx.x >> 5;
  const int lo = lane & 15, hi = lane >> 4;
  const int q0 = (blockIdx.x * 4 + wid) * 16;
  const int bh = blockIdx.y;
  const bf16* Qb = Qs + (size_t)bh * S_LEN * HDPAD;
  const bf16* Kb = Ks + (size_t)bh * S_LEN * HDPAD;
  const bf16* Vb = Vt + (size_t)bh * HDIM * S_LEN;

  v16bf aq[3];
#pragma unroll
  for (int i = 0; i < 3; ++i)
    aq[i] = frag_a(Qb + (size_t)(q0 + lo) * HDPAD + 32 * i, hi);

  float mrow[8], lrow[8];
#pragma unroll
  for (int r = 0; r < 8; ++r) { mrow[r] = -1e30f; lrow[r] = 0.f; }
  v8f o[5] = {};

  const int nkt = (q0 + 47) / 32;  // key tiles covering keys <= q0+15
  for (int kt = 0; kt < nkt; ++kt) {
    const int key0 = kt * 32;
    if (kt + 1 < nkt)
      __builtin_prefetch(Kb + (size_t)(key0 + 32 + lane) * HDPAD, 0, 0);

    // S = Q . K^T  (2 n-tiles of 16 keys, K-loop 3 x 32 over padded head dim)
    v8f sc[2] = {};
#pragma unroll
    for (int i = 0; i < 3; ++i) {
      v16bf bk0 = frag_b(Kb + (size_t)(key0 + lo) * HDPAD + 32 * i, hi);
      v16bf bk1 = frag_b(Kb + (size_t)(key0 + 16 + lo) * HDPAD + 32 * i, hi);
      sc[0] = wmma_bf16(aq[i], bk0, sc[0]);
      sc[1] = wmma_bf16(aq[i], bk1, sc[1]);
    }
    if (key0 + 31 > q0) {  // causal mask (== reference additive -1e9 mask)
#pragma unroll
      for (int t = 0; t < 2; ++t)
#pragma unroll
        for (int v = 0; v < 8; ++v) {
          int key = key0 + t * 16 + lo;
          int q = q0 + v + hi * 8;
          if (key > q) sc[t][v] = -1e30f;
        }
    }
    // online softmax; row r of frag = q row (r + hi*8); cols live across lanes
#pragma unroll
    for (int r = 0; r < 8; ++r) {
      float s0 = sc[0][r], s1 = sc[1][r];
      float mx = fmaxf(s0, s1);
      mx = fmaxf(mx, __shfl_xor(mx, 1, 32));
      mx = fmaxf(mx, __shfl_xor(mx, 2, 32));
      mx = fmaxf(mx, __shfl_xor(mx, 4, 32));
      mx = fmaxf(mx, __shfl_xor(mx, 8, 32));
      float mn = fmaxf(mrow[r], mx);
      float corr = __expf(mrow[r] - mn);
      float p0 = __expf(s0 - mn), p1 = __expf(s1 - mn);
      float rs = p0 + p1;
      rs += __shfl_xor(rs, 1, 32);
      rs += __shfl_xor(rs, 2, 32);
      rs += __shfl_xor(rs, 4, 32);
      rs += __shfl_xor(rs, 8, 32);
      lrow[r] = lrow[r] * corr + rs;
      mrow[r] = mn;
      sc[0][r] = p0; sc[1][r] = p1;
#pragma unroll
      for (int t = 0; t < 5; ++t) o[t][r] *= corr;
    }
    // reshape P: C layout -> A layout through wave-private LDS (wave-local dep)
    bf16* pb = &lP[wid][0];
#pragma unroll
    for (int t = 0; t < 2; ++t)
#pragma unroll
      for (int v = 0; v < 8; ++v)
        pb[(v + hi * 8) * 32 + t * 16 + lo] = (bf16)sc[t][v];
    asm volatile("s_wait_dscnt 0" ::: "memory");
    v16bf aP = frag_a(&pb[lo * 32], hi);
    // O += P . V  (V stored [d][s]: B-fragment is two runs of 8 contiguous keys)
#pragma unroll
    for (int t = 0; t < 5; ++t) {
      v16bf bv = frag_b(Vb + (size_t)(t * 16 + lo) * S_LEN + key0, hi);
      o[t] = wmma_bf16(aP, bv, o[t]);
    }
  }

  const int b = bh >> 4, h = bh & 15;
#pragma unroll
  for (int v = 0; v < 8; ++v) {
    float inv = 1.f / lrow[v];
    int q = q0 + v + hi * 8;
#pragma unroll
    for (int t = 0; t < 5; ++t) {
      int d = t * 16 + lo;
      Oa[((size_t)(b * S_LEN + q)) * DMODEL + h * HDIM + d] =
          (bf16)(o[t][v] * inv);
    }
  }
}

// ---------------- Output projection: out = A @ o_w^T + o_b (f32 out) -------
__global__ __launch_bounds__(256) void k_oproj_gemm(
    const bf16* __restrict__ A,     // [MTOT][DMODEL]
    const bf16* __restrict__ W,     // [DMODEL][DMODEL]
    const float* __restrict__ bias, // [DMODEL]
    float* __restrict__ out)        // [MTOT][DMODEL]
{
  __shared__ bf16 lA[64 * 32];
  __shared__ bf16 lB[128 * 32];
  const int tid = threadIdx.x;
  const int lane = tid & 31, wid = tid >> 5;
  const int lo = lane & 15, hi = lane >> 4;
  const int wr = wid >> 2, wc = wid & 3;
  const int m0 = blockIdx.y * 64;
  const int n0 = blockIdx.x * 128;

  const int mL = tid >> 2, segA = tid & 3;
  const int nL0 = tid >> 2,         segB0 = tid & 3;
  const int nL1 = (tid + 256) >> 2, segB1 = (tid + 256) & 3;
  const bf16* gA  = &A[(size_t)(m0 + mL) * DMODEL + segA * 8];
  const bf16* gB0 = &W[(size_t)(n0 + nL0) * DMODEL + segB0 * 8];
  const bf16* gB1 = &W[(size_t)(n0 + nL1) * DMODEL + segB1 * 8];

  v8f c[2][2] = {};
  v8bf ra  = *(const v8bf*)gA;
  v8bf rb0 = *(const v8bf*)gB0;
  v8bf rb1 = *(const v8bf*)gB1;

  for (int k0 = 0; k0 < DMODEL; k0 += 32) {
    *(v8bf*)&lA[mL * 32 + segA * 8]   = ra;
    *(v8bf*)&lB[nL0 * 32 + segB0 * 8] = rb0;
    *(v8bf*)&lB[nL1 * 32 + segB1 * 8] = rb1;
    __syncthreads();
    if (k0 + 32 < DMODEL) {
      ra  = *(const v8bf*)(gA + k0 + 32);
      rb0 = *(const v8bf*)(gB0 + k0 + 32);
      rb1 = *(const v8bf*)(gB1 + k0 + 32);
    }
    v16bf a0 = frag_a(&lA[(wr * 32 + 0 * 16 + lo) * 32], hi);
    v16bf a1 = frag_a(&lA[(wr * 32 + 1 * 16 + lo) * 32], hi);
    v16bf b0 = frag_b(&lB[(wc * 32 + 0 * 16 + lo) * 32], hi);
    v16bf b1 = frag_b(&lB[(wc * 32 + 1 * 16 + lo) * 32], hi);
    c[0][0] = wmma_bf16(a0, b0, c[0][0]);
    c[0][1] = wmma_bf16(a0, b1, c[0][1]);
    c[1][0] = wmma_bf16(a1, b0, c[1][0]);
    c[1][1] = wmma_bf16(a1, b1, c[1][1]);
    __syncthreads();
  }
#pragma unroll
  for (int i = 0; i < 2; ++i)
#pragma unroll
    for (int j = 0; j < 2; ++j)
#pragma unroll
      for (int v = 0; v < 8; ++v) {
        int m = m0 + wr * 32 + i * 16 + v + hi * 8;
        int n = n0 + wc * 32 + j * 16 + lo;
        out[(size_t)m * DMODEL + n] = c[i][j][v] + bias[n];
      }
}

extern "C" void kernel_launch(void* const* d_in, const int* in_sizes, int n_in,
                              void* d_out, int out_size, void* d_ws,
                              size_t ws_size, hipStream_t stream) {
  (void)in_sizes; (void)n_in; (void)out_size; (void)ws_size;
  const float* hs   = (const float*)d_in[0];
  // d_in[1] = attention_mask (causal additive mask) -> implemented analytically
  const float* scal = (const float*)d_in[2];
  const float* qkvw = (const float*)d_in[3];
  const float* qkvb = (const float*)d_in[4];
  const float* ow   = (const float*)d_in[5];
  const float* ob   = (const float*)d_in[6];
  float* out = (float*)d_out;

  char* ws = (char*)d_ws;
  bf16* Qs = (bf16*)ws; ws += (size_t)BATCH * NHEAD * S_LEN * HDPAD * 2;
  bf16* Ks = (bf16*)ws; ws += (size_t)BATCH * NHEAD * S_LEN * HDPAD * 2;
  bf16* Vt = (bf16*)ws; ws += (size_t)BATCH * NHEAD * HDIM * S_LEN * 2;
  bf16* Wq = (bf16*)ws; ws += (size_t)NQKV * DMODEL * 2;
  bf16* Wo = (bf16*)ws; ws += (size_t)DMODEL * DMODEL * 2;
  bf16* Hs = (bf16*)ws;  // hidden bf16; region reused for attention output

  k_cvt_bf16<<<1024, 256, 0, stream>>>(hs, Hs, MTOT * DMODEL);
  k_cvt_bf16<<<1024, 256, 0, stream>>>(qkvw, Wq, NQKV * DMODEL);
  k_cvt_bf16<<<512, 256, 0, stream>>>(ow, Wo, DMODEL * DMODEL);

  int n8 = 2 * BATCH * NHEAD * S_LEN * HDPAD / 8;  // Q and K are contiguous
  k_zero_bf16<<<(n8 + 255) / 256, 256, 0, stream>>>(Qs, n8);

  k_qkv_gemm<<<dim3(NQKV / 128, MTOT / 64), 256, 0, stream>>>(
      Hs, Wq, qkvb, scal, Qs, Ks, Vt);
  k_flash_attn<<<dim3(S_LEN / 64, BATCH * NHEAD), 128, 0, stream>>>(
      Qs, Ks, Vt, Hs);
  k_oproj_gemm<<<dim3(DMODEL / 128, MTOT / 64), 256, 0, stream>>>(
      Hs, Wo, ob, out);
}
